// SACConv_8143257993620
// MI455X (gfx1250) — matrix operand order repaired and checked by reading, not compiled, not run on hardware
//
#include <hip/hip_runtime.h>

typedef __attribute__((ext_vector_type(16))) _Float16 v16h;
typedef __attribute__((ext_vector_type(8)))  float    v8f;

#define C1    16
#define C2    16
#define HDIM  16
#define HH    64
#define WW    64
#define NPE   144          // C1*3*3 patch length
#define NT    17           // n-tiles: 16 channels + 1 bias tile
#define NKC   5            // K chunks of 32 (144 padded to 160)
#define TW    16           // tile width  (x)
#define THT   8            // tile height (y)
#define NLOC  128          // locations per workgroup
#define NTHR  128          // threads per workgroup (4 waves)
#define XT_W  18
#define XT_H  10
#define XT_C  (XT_W*XT_H)  // 180 floats per channel plane

// single-instruction f32x2 -> packed f16x2 (v_cvt_pk_rtz_f16_f32)
static __device__ __forceinline__ unsigned pack2(float a, float b) {
    return __builtin_bit_cast(unsigned, __builtin_amdgcn_cvt_pkrtz(a, b));
}

union Frag { unsigned u[8]; v16h h; };

// ---------------------------------------------------------------------------
// Prep: repack [fc2_w | fc2_b] into wave32 WMMA B-fragment order in d_ws.
// Linear dword index = ((t*NKC + kc)*32 + lane)*8 + r
// B k order used here (contiguous, same as A): VGPR r, lanes 0-15 (N=lane):     K = 2r, 2r+1
//                             lanes 16-31 (N=lane-16): K = 16+2r, 17+2r
// n = t*16 + (lane&15);  n<256 -> (c=n>>4, j=n&15) weight col; n>=256 -> bias c
// ---------------------------------------------------------------------------
__global__ void sac_prep_b(const float* __restrict__ fc2_w,
                           const float* __restrict__ fc2_b,
                           unsigned* __restrict__ wsB) {
    int idx = blockIdx.x * 256 + threadIdx.x;          // 85*256 == NT*NKC*32*8
    int t    = idx / (NKC * 32 * 8);
    int rem  = idx % (NKC * 32 * 8);
    int kc   = rem / (32 * 8);
    int rem2 = rem % (32 * 8);
    int lane = rem2 >> 3;
    int r    = rem2 & 7;
    int n = t * 16 + (lane & 15);
    int k = kc * 32 + 2 * r + ((lane >= 16) ? 16 : 0);
    float f0 = 0.f, f1 = 0.f;
    if (k < NPE) {
        if (n < 256) {
            int c = n >> 4, j = n & 15;
            f0 = fc2_w[(c * NPE + k    ) * HDIM + j];
            f1 = fc2_w[(c * NPE + k + 1) * HDIM + j];
        } else {
            int c = n - 256;
            f0 = fc2_b[c * NPE + k];
            f1 = fc2_b[c * NPE + k + 1];
        }
    }
    wsB[idx] = pack2(f0, f1);
}

// ---------------------------------------------------------------------------
// Main kernel: one workgroup = 16x8 output locations of one batch image.
// ---------------------------------------------------------------------------
__launch_bounds__(NTHR)
__global__ void sac_main(const float* __restrict__ x,
                         const float* __restrict__ fc1_w,
                         const float* __restrict__ fc1_b,
                         const unsigned* __restrict__ wsB,
                         float* __restrict__ out) {
    __shared__ float xt[C1 * XT_C];       // input tile + halo (f32)
    __shared__ int   offs[160];           // o -> channel-plane offset LUT
    __shared__ float hl[NLOC * HDIM];     // per-location MLP hidden
    __shared__ float outacc[4 * 256];     // per-wave [lx][c] accumulator
    __shared__ float dscr[4 * 256];       // per-wave WMMA D staging

    const int tid = threadIdx.x;
    const int tx = blockIdx.x, ty = blockIdx.y, b = blockIdx.z;

    // ---- stage input tile (zero-padded 'same' halo) ----
    for (int i = tid; i < C1 * XT_C; i += NTHR) {
        int c  = i / XT_C;  int r  = i - c * XT_C;
        int yy = r / XT_W;  int xx = r - yy * XT_W;
        int gy = ty * THT + yy - 1, gx = tx * TW + xx - 1;
        float v = 0.f;
        if (gy >= 0 && gy < HH && gx >= 0 && gx < WW)
            v = x[(((b * C1 + c) * HH) + gy) * WW + gx];
        xt[i] = v;
    }
    if (tid < 160) {
        int o = tid;
        int c = o / 9, rm = o - c * 9, i2 = rm / 3, j2 = rm - i2 * 3;
        offs[tid] = (o < NPE) ? (c * XT_C + i2 * XT_W + j2) : 0;
    }
    __syncthreads();

    // ---- stats + fc1: one location per thread (VALU) ----
    {
        const int ly = tid >> 4, lx = tid & 15;
        const int base = ly * XT_W + lx;
        float s1 = 0.f, s2 = 0.f, mx = -3.4e38f;
        for (int c = 0; c < C1; ++c) {
#pragma unroll
            for (int ij = 0; ij < 9; ++ij) {
                float v = xt[c * XT_C + (ij / 3) * XT_W + (ij % 3) + base];
                s1 += v; s2 += v * v; mx = fmaxf(mx, v);
            }
        }
        const float invN = 1.f / 144.f;
        float mu = s1 * invN;
        float vs = fmaxf(s2 - 144.f * mu * mu, 0.f);
        float sigma = sqrtf(vs * (1.f / 143.f)) + 1e-6f;
        float s3 = 0.f, s4 = 0.f, S = 0.f;
        for (int c = 0; c < C1; ++c) {
#pragma unroll
            for (int ij = 0; ij < 9; ++ij) {
                float v = xt[c * XT_C + (ij / 3) * XT_W + (ij % 3) + base];
                float d = v - mu, d2 = d * d;
                s3 += d2 * d; s4 += d2 * d2;
                S += __expf(v - mx);
            }
        }
        float invS = 1.f / S;
        float ent = 0.f;
        for (int c = 0; c < C1; ++c) {
#pragma unroll
            for (int ij = 0; ij < 9; ++ij) {
                float v = xt[c * XT_C + (ij / 3) * XT_W + (ij % 3) + base];
                float p = __expf(v - mx) * invS;
                ent -= p * __logf(p + 1e-6f);
            }
        }
        float is = 1.f / sigma;
        float st[5];
        st[0] = mu; st[1] = sigma;
        st[2] = s3 * invN * is * is * is;
        st[3] = s4 * invN * is * is * is * is - 3.f;
        st[4] = ent;
#pragma unroll
        for (int j = 0; j < HDIM; ++j) {
            float a = fc1_b[j];
#pragma unroll
            for (int m = 0; m < 5; ++m) a += fc1_w[j * 5 + m] * st[m];
            hl[tid * HDIM + j] = fmaxf(a, 0.f);
        }
    }
    __syncthreads();

    // ---- WMMA phase: wave w handles l-tiles (rows) 2w, 2w+1 ----
    const int w    = tid >> 5;
    const int lane = tid & 31;
    const int half = lane >> 4;
    const int lx   = lane & 15;
    const uint4* wsq = (const uint4*)wsB;

    for (int lt2 = 0; lt2 < 2; ++lt2) {
        const int lt = w * 2 + lt2;                 // row within 8-row tile
        // A fragments: patches for 16 locations (M=lx), f16, K padded to 160
        // A k order used here (contiguous, same as B): K = 2r + (half?16:0)
        Frag a[NKC];
        const int abase = lt * XT_W + lx;
#pragma unroll
        for (int kc = 0; kc < NKC; ++kc) {
#pragma unroll
            for (int r = 0; r < 8; ++r) {
                int k0 = kc * 32 + 2 * r + (half ? 16 : 0);
                float f0 = 0.f, f1 = 0.f;
                if (k0 < NPE) {                      // k0 even -> k0+1 <= 143
                    f0 = xt[offs[k0]     + abase];
                    f1 = xt[offs[k0 + 1] + abase];
                }
                a[kc].u[r] = pack2(f0, f1);
            }
        }
        // cache hidden vector for this lane's location (dup across halves)
        float hreg[HDIM];
#pragma unroll
        for (int j = 0; j < HDIM; ++j) hreg[j] = hl[(lt * 16 + lx) * HDIM + j];

        const int rowbase = w * 256 + (half ? 128 : 0) + lx;  // D row v+8*half, col lx

        // bias n-tile (t=16) first: its D tile IS the [lx][c] output base
        {
            v8f acc = {};
#pragma unroll
            for (int kc = 0; kc < NKC; ++kc) {
                int bi = ((16 * NKC + kc) * 32 + lane) * 2;
                uint4 q0 = wsq[bi], q1 = wsq[bi + 1];
                Frag bu;
                bu.u[0] = q0.x; bu.u[1] = q0.y; bu.u[2] = q0.z; bu.u[3] = q0.w;
                bu.u[4] = q1.x; bu.u[5] = q1.y; bu.u[6] = q1.z; bu.u[7] = q1.w;
                acc = __builtin_amdgcn_wmma_f32_16x16x32_f16(
                        false, a[kc].h, false, bu.h, (short)0, acc, false, false);
            }
#pragma unroll
            for (int v = 0; v < 8; ++v) outacc[rowbase + v * 16] = acc[v];
        }
        asm volatile("s_wait_dscnt 0" ::: "memory");

#pragma unroll 2
        for (int t = 0; t < 16; ++t) {               // t == output channel c
            v8f acc = {};
#pragma unroll
            for (int kc = 0; kc < NKC; ++kc) {
                int bi = ((t * NKC + kc) * 32 + lane) * 2;
                uint4 q0 = wsq[bi], q1 = wsq[bi + 1];
                Frag bu;
                bu.u[0] = q0.x; bu.u[1] = q0.y; bu.u[2] = q0.z; bu.u[3] = q0.w;
                bu.u[4] = q1.x; bu.u[5] = q1.y; bu.u[6] = q1.z; bu.u[7] = q1.w;
                acc = __builtin_amdgcn_wmma_f32_16x16x32_f16(
                        false, a[kc].h, false, bu.h, (short)0, acc, false, false);
            }
#pragma unroll
            for (int v = 0; v < 8; ++v) dscr[rowbase + v * 16] = acc[v];
            asm volatile("s_wait_dscnt 0" ::: "memory");
            if (lane < 16) {                         // out[l=lane, c=t] += h . M
                float s = 0.f;
#pragma unroll
                for (int j = 0; j < HDIM; ++j)
                    s += hreg[j] * dscr[w * 256 + lane * 16 + j];
                outacc[w * 256 + lane * 16 + t] += s;
            }
            asm volatile("s_wait_dscnt 0" ::: "memory");
        }

        // ---- write out: [B, C2, 64, 64], coalesced over gx ----
        {
            const int gy = ty * THT + lt;
            const int gxb = tx * TW;
#pragma unroll
            for (int i = 0; i < 8; ++i) {
                int c = i * 2 + half;
                float v = outacc[w * 256 + lx * 16 + c];
                out[(((b * C2 + c) * HH) + gy) * WW + gxb + lx] = v;
            }
        }
        asm volatile("s_wait_dscnt 0" ::: "memory");
    }
}

extern "C" void kernel_launch(void* const* d_in, const int* in_sizes, int n_in,
                              void* d_out, int out_size, void* d_ws, size_t ws_size,
                              hipStream_t stream) {
    const float* x     = (const float*)d_in[0];
    const float* fc1_w = (const float*)d_in[1];
    const float* fc1_b = (const float*)d_in[2];
    const float* fc2_w = (const float*)d_in[3];
    const float* fc2_b = (const float*)d_in[4];
    float* out = (float*)d_out;
    unsigned* wsB = (unsigned*)d_ws;                 // needs 17*5*32*8*4 = 87,040 B

    hipLaunchKernelGGL(sac_prep_b, dim3(85), dim3(256), 0, stream,
                       fc2_w, fc2_b, wsB);
    hipLaunchKernelGGL(sac_main, dim3(4, 8, 8), dim3(NTHR), 0, stream,
                       x, fc1_w, fc1_b, wsB, out);
}
